// VQVAELayer_39573828665698
// MI455X (gfx1250) — matrix operand order repaired and checked by reading, not compiled, not run on hardware
//
#include <hip/hip_runtime.h>
#include <hip/hip_bf16.h>

typedef __attribute__((ext_vector_type(16))) __bf16 v16bf;
typedef __attribute__((ext_vector_type(8)))  __bf16 v8bf;
typedef __attribute__((ext_vector_type(8)))  float  v8f;
typedef unsigned int u32x4 __attribute__((ext_vector_type(4)));
typedef int          i32x8 __attribute__((ext_vector_type(8)));
typedef int          i32x4 __attribute__((ext_vector_type(4)));

#define EMBED_DIM 256
#define NUM_EMBED 4096
#define N_ROWS    16384
#define STAGE_CODES 128            // codes staged in LDS per stage (64KB of B data)
#define B_STRIDE   264             // 256 + 8 halves pad -> 4-dword bank skew per code row
#define NSTAGE     (NUM_EMBED / STAGE_CODES)
#define STAGE_GBYTES (STAGE_CODES * EMBED_DIM * 2)   // bf16 bytes per stage in global
#define STAGE_LHALVES (STAGE_CODES * B_STRIDE)       // halves per LDS buffer

// ---------------------------------------------------------------------------
// Kernel 1: transpose w [D=256, K=4096] -> wT [K, D] in bf16 (WMMA B operand)
// and fp32 (exact gather source). LDS-tiled 32x32 for coalescing both sides.
// ---------------------------------------------------------------------------
__global__ __launch_bounds__(256) void k_transpose(const float* __restrict__ w,
                                                   __bf16* __restrict__ wTb,
                                                   float*  __restrict__ wTf) {
  __shared__ float t[32][33];
  const int k0 = blockIdx.x * 32;
  const int d0 = blockIdx.y * 32;
  const int tx = threadIdx.x;      // 0..31
  const int ty = threadIdx.y;      // 0..7
#pragma unroll
  for (int i = 0; i < 4; ++i) {
    const int dl = ty + i * 8;
    t[dl][tx] = w[(size_t)(d0 + dl) * NUM_EMBED + k0 + tx];   // coalesced in k
  }
  __syncthreads();
#pragma unroll
  for (int i = 0; i < 4; ++i) {
    const int kl = ty + i * 8;
    const float v = t[tx][kl];
    const size_t o = (size_t)(k0 + kl) * EMBED_DIM + d0 + tx; // coalesced in d
    wTf[o] = v;
    wTb[o] = (__bf16)v;
  }
}

// ---------------------------------------------------------------------------
// Kernel 2: wn[k] = 0.5 * ||w_k||^2 ; one wave32 per code, contiguous reads.
// ---------------------------------------------------------------------------
__global__ __launch_bounds__(256) void k_wn(const float* __restrict__ wTf,
                                            float* __restrict__ wn) {
  const int wave = threadIdx.x >> 5;
  const int lane = threadIdx.x & 31;
  const int k = blockIdx.x * 8 + wave;
  const float4* p = (const float4*)(wTf + (size_t)k * EMBED_DIM);
  float s = 0.f;
#pragma unroll
  for (int i = 0; i < 2; ++i) {
    const float4 v = p[lane * 2 + i];
    s += v.x * v.x + v.y * v.y + v.z * v.z + v.w * v.w;
  }
#pragma unroll
  for (int m = 16; m >= 1; m >>= 1) s += __shfl_xor(s, m, 32);
  if (lane == 0) wn[k] = 0.5f * s;
}

// ---------------------------------------------------------------------------
// Kernel 3: argmin via bf16 WMMA.
//  - 128 threads (4 waves), each wave owns 32 rows = two 16-row A tiles kept
//    resident in VGPRs -> two independent WMMA accumulator chains (2-way ILP,
//    2 WMMAs per B-fragment LDS read).
//  - B staged to LDS by the Tensor Data Mover (wave 0 issues one
//    tensor_load_to_lds per 64KB stage; D# pad fields insert the 16B bank
//    skew), double-buffered against compute; s_wait_tensorcnt + barrier.
//  - Score = dot(x,w_k) - 0.5||w_k||^2 ; argmax == argmin distance.
// ---------------------------------------------------------------------------
__global__ __launch_bounds__(128) void k_argmin(const float*  __restrict__ x,
                                                const __bf16* __restrict__ wTb,
                                                const float*  __restrict__ wn,
                                                int*          __restrict__ idxOut,
                                                float*        __restrict__ fIdxOut) {
  __shared__ __bf16 Bs[2][STAGE_LHALVES];

  const int tid  = threadIdx.x;
  const int wave = tid >> 5;
  const int lane = tid & 31;
  const int lr   = lane & 15;   // column / row-in-tile lane
  const int hi   = lane >> 4;   // half-wave id
  const int m0   = blockIdx.x * 128 + wave * 32;

  // --- Preload A fragments for both 16-row tiles (ISA 16-bit A 16x32 layout:
  // lanes 0-15: K = base+0..7 / base+16..23 ; lanes 16-31: +8).
  v16bf a0[8], a1[8];
  {
    const float* xr0 = x + (size_t)(m0 + lr) * EMBED_DIM;
    const float* xr1 = x + (size_t)(m0 + 16 + lr) * EMBED_DIM;
#pragma unroll
    for (int dc = 0; dc < 8; ++dc) {
      const int b0 = dc * 32 + hi * 8;
#pragma unroll
      for (int i = 0; i < 8; ++i) {
        a0[dc][i]     = (__bf16)xr0[b0 + i];
        a0[dc][8 + i] = (__bf16)xr0[b0 + 16 + i];
        a1[dc][i]     = (__bf16)xr1[b0 + i];
        a1[dc][8 + i] = (__bf16)xr1[b0 + 16 + i];
      }
    }
  }

  float bv0[8], bv1[8];
  int   bi0[8], bi1[8];
#pragma unroll
  for (int j = 0; j < 8; ++j) {
    bv0[j] = -3.402823466e38f; bi0[j] = 0;
    bv1[j] = -3.402823466e38f; bi1[j] = 0;
  }

  const unsigned long long gbase = (unsigned long long)(size_t)wTb;
  const unsigned ldsBase = (unsigned)(size_t)(&Bs[0][0]);

  // TDM descriptor issue: 2D tile, 128 dwords/row x 128 rows, contiguous in
  // global, LDS padded by 4 dwords every 128 dwords (pad_interval=6, amount=3).
  auto tdm_issue = [&](int st, int buf) {
    const unsigned long long ga = gbase + (unsigned long long)st * STAGE_GBYTES;
    const unsigned laddr = ldsBase + (unsigned)buf * (STAGE_LHALVES * 2);
    u32x4 g0;
    g0[0] = 1u;                                       // count=1, user mode
    g0[1] = laddr;                                    // lds_addr (bytes)
    g0[2] = (unsigned)(ga & 0xffffffffu);             // global_addr[31:0]
    g0[3] = (unsigned)((ga >> 32) & 0x01ffffffu)      // global_addr[56:32]
          | 0x80000000u;                              // type=2 ("image")
    i32x8 g1;
    g1[0] = (int)((2u << 16)     // data_size = 4B
                | (1u << 20)     // pad_enable
                | (6u << 22)     // pad_interval: 128 dwords
                | (3u << 25));   // pad_amount: 4 dwords
    g1[1] = (int)(128u << 16);   // tensor_dim0[15:0]=128 (atomic_barrier=0)
    g1[2] = (int)(128u << 16);   // tensor_dim0 hi=0 | tensor_dim1[15:0]=128
    g1[3] = (int)(128u << 16);   // tensor_dim1 hi=0 | tile_dim0=128
    g1[4] = (int)(128u);         // tile_dim1=128 | tile_dim2=0
    g1[5] = (int)(128u);         // tensor_dim0_stride lo32 = 128 dwords
    g1[6] = 0;                   // stride hi | dim1_stride lo (unused, 2D)
    g1[7] = 0;
    const i32x4 gz4 = {0, 0, 0, 0};
    const i32x8 gz8 = {0, 0, 0, 0, 0, 0, 0, 0};
    __builtin_amdgcn_tensor_load_to_lds(g0, g1, gz4, gz4, gz8, 0);
  };

  // Prologue: stage 0 into buffer 0.
  if (wave == 0) {
    tdm_issue(0, 0);
    __builtin_amdgcn_s_wait_tensorcnt(0);
  }
  __syncthreads();

  for (int st = 0; st < NSTAGE; ++st) {
    const int cur = st & 1;
    if (wave == 0 && st + 1 < NSTAGE) tdm_issue(st + 1, cur ^ 1);  // prefetch next stage

    const int n0 = st * STAGE_CODES;
    const __bf16* bufp = &Bs[cur][0];

    for (int s = 0; s < 8; ++s) {
      const int  nb   = n0 + s * 16;
      const int  code = nb + lr;
      const float wnv = wn[code];                 // one code per lane in C-layout
      const __bf16* bp = bufp + (s * 16 + lr) * B_STRIDE + hi * 8;

      // Software-pipelined B fragments: load dc+1 while WMMA consumes dc.
      v8bf blo = *(const v8bf*)(bp);
      v8bf bhi = *(const v8bf*)(bp + 16);
      v8f c0 = {0.f, 0.f, 0.f, 0.f, 0.f, 0.f, 0.f, 0.f};
      v8f c1 = {0.f, 0.f, 0.f, 0.f, 0.f, 0.f, 0.f, 0.f};
#pragma unroll
      for (int dc = 0; dc < 8; ++dc) {
        const v16bf b = __builtin_shufflevector(blo, bhi,
            0, 1, 2, 3, 4, 5, 6, 7, 8, 9, 10, 11, 12, 13, 14, 15);
        if (dc < 7) {
          blo = *(const v8bf*)(bp + (dc + 1) * 32);
          bhi = *(const v8bf*)(bp + (dc + 1) * 32 + 16);
        }
        c0 = __builtin_amdgcn_wmma_f32_16x16x32_bf16(
            false, a0[dc], false, b, (short)0, c0, false, false);
        c1 = __builtin_amdgcn_wmma_f32_16x16x32_bf16(
            false, a1[dc], false, b, (short)0, c1, false, false);
      }
      // C layout: VGPR j -> (M = j + 8*hi, N = lr): all 8 results share a code.
#pragma unroll
      for (int j = 0; j < 8; ++j) {
        const float s0 = c0[j] - wnv;
        const float s1 = c1[j] - wnv;
        if (s0 > bv0[j]) { bv0[j] = s0; bi0[j] = code; }  // ascending => min idx on tie
        if (s1 > bv1[j]) { bv1[j] = s1; bi1[j] = code; }
      }
    }

    __syncthreads();                                   // all waves done with buf[cur]
    if (wave == 0 && st + 1 < NSTAGE)
      __builtin_amdgcn_s_wait_tensorcnt(0);            // next stage landed
    __syncthreads();                                   // publish buf[cur^1]
  }

  // Reduce best (val,idx) across the 16 lanes of each half-wave (ties -> min idx).
#pragma unroll
  for (int m = 1; m < 16; m <<= 1) {
#pragma unroll
    for (int j = 0; j < 8; ++j) {
      float ov; int oi;
      ov = __shfl_xor(bv0[j], m, 32); oi = __shfl_xor(bi0[j], m, 32);
      if (ov > bv0[j] || (ov == bv0[j] && oi < bi0[j])) { bv0[j] = ov; bi0[j] = oi; }
      ov = __shfl_xor(bv1[j], m, 32); oi = __shfl_xor(bi1[j], m, 32);
      if (ov > bv1[j] || (ov == bv1[j] && oi < bi1[j])) { bv1[j] = ov; bi1[j] = oi; }
    }
  }
  if (lr == 0) {
#pragma unroll
    for (int j = 0; j < 8; ++j) {
      const int r0 = m0 + hi * 8 + j;
      const int r1 = m0 + 16 + hi * 8 + j;
      idxOut[r0]  = bi0[j];  fIdxOut[r0] = (float)bi0[j];
      idxOut[r1]  = bi1[j];  fIdxOut[r1] = (float)bi1[j];
    }
  }
}

// ---------------------------------------------------------------------------
// Kernel 4: exact fp32 gather of the winning codebook rows (coalesced).
// ---------------------------------------------------------------------------
__global__ __launch_bounds__(64) void k_gather(const int*   __restrict__ idx,
                                               const float* __restrict__ wTf,
                                               float*       __restrict__ outQ) {
  const int row = blockIdx.x;
  const int t   = threadIdx.x;  // 0..63, float4 each
  const int id  = idx[row];
  const float4* s = (const float4*)(wTf + (size_t)id * EMBED_DIM);
  ((float4*)(outQ + (size_t)row * EMBED_DIM))[t] = s[t];
}

// ---------------------------------------------------------------------------
extern "C" void kernel_launch(void* const* d_in, const int* in_sizes, int n_in,
                              void* d_out, int out_size, void* d_ws, size_t ws_size,
                              hipStream_t stream) {
  const float* x = (const float*)d_in[0];   // [16,1024,256] fp32
  const float* w = (const float*)d_in[1];   // [256,4096]    fp32

  float* out  = (float*)d_out;
  float* fIdx = out;              // indices (as float), 16384 elements
  float* outQ = out + N_ROWS;     // quantized, 16384*256 fp32

  char* ws = (char*)d_ws;
  __bf16* wTb = (__bf16*)ws;                              // 2 MB  : wT bf16 [K,D]
  float*  wTf = (float*)(ws + (2u << 20));                // 4 MB  : wT fp32 [K,D]
  float*  wn  = (float*)(ws + (6u << 20));                // 16 KB : 0.5*||w_k||^2
  int*    idx = (int*)  (ws + (6u << 20) + (64u << 10));  // 64 KB : argmin indices

  k_transpose<<<dim3(NUM_EMBED / 32, EMBED_DIM / 32), dim3(32, 8), 0, stream>>>(w, wTb, wTf);
  k_wn<<<NUM_EMBED / 8, 256, 0, stream>>>(wTf, wn);
  k_argmin<<<N_ROWS / 128, 128, 0, stream>>>(x, wTb, wn, idx, fIdx);
  k_gather<<<N_ROWS, 64, 0, stream>>>(idx, wTf, outQ);
}
